// ErbStage_78254304133851
// MI455X (gfx1250) — compile-verified, matrix operand order checked by reading, not run on hardware
//
#include <hip/hip_runtime.h>
#include <hip/hip_bf16.h>
#include <math.h>

// ---------------------------------------------------------------------------
// MI455X (gfx1250) implementation. wave32. Matrix work via v_wmma_f32_16x16x32_f16.
// ---------------------------------------------------------------------------

typedef _Float16 v16h __attribute__((ext_vector_type(16)));
typedef _Float16 v8h  __attribute__((ext_vector_type(8)));
typedef float    v8f  __attribute__((ext_vector_type(8)));

#define B_   32
#define T_   1000
#define F_   32
#define C0_  16
#define C1_  8
#define H_   64
#define DH   256          // C1*F  (GRU hidden)
#define G3   768          // 3*DH
#define DIN  2048         // F*H   (GRU input)
#define M_   (B_*T_)      // 32000 rows

__device__ __forceinline__ float sigmoidf_(float x) { return 1.0f / (1.0f + expf(-x)); }

// Per-lane fragment load for V_WMMA_F32_16X16X32_F16 operands.
// ISA 16-bit A layout (16x32, M x K): lane = row (mod 16), within a lane the
// 16 halfs are K = {8*(lane/16) .. +7} and {8*(lane/16)+16 .. +23}  -> two
// contiguous 16-byte chunks at element offsets kbase and kbase+16.  The B
// operand is B^T rows in the same per-lane layout; our weights are [N,K]
// row-major so they feed SRC1 directly through this helper.
__device__ __forceinline__ v16h ldfrag(const _Float16* p) {
  v8h lo = *(const v8h*)(p);
  v8h hi = *(const v8h*)(p + 16);
  v16h r;
#pragma unroll
  for (int i = 0; i < 8; ++i) { r[i] = lo[i]; r[i + 8] = hi[i]; }
  return r;
}

#define WMMA_F16(a, b, c) \
  __builtin_amdgcn_wmma_f32_16x16x32_f16(false, (a), false, (b), (short)0, (c), false, false)

// ---------------------------------------------------------------------------
// Kernel 1: fused frontend per (b,t).  conv(1,3) acts on freq only -> no time
// context, so each (b,t) is independent.
// ---------------------------------------------------------------------------
__global__ __launch_bounds__(256)
void frontend_kernel(const float* __restrict__ in,
                     const float* __restrict__ w0, const float* __restrict__ b0,
                     const float* __restrict__ g0, const float* __restrict__ be0,
                     const float* __restrict__ w1, const float* __restrict__ b1,
                     const float* __restrict__ g1, const float* __restrict__ be1,
                     const float* __restrict__ qw, const float* __restrict__ qb,
                     const float* __restrict__ kw, const float* __restrict__ kb,
                     const float* __restrict__ vw, const float* __restrict__ vb,
                     float* __restrict__ x1out, _Float16* __restrict__ gi) {
  __shared__ float s_in[F_];
  __shared__ float s_x0[C0_ * F_];
  __shared__ float s_x1[C1_ * F_];
  __shared__ float s_q[H_ * F_];
  __shared__ float s_k[H_ * F_];
  __shared__ float s_v[H_ * F_];
  __shared__ float s_w[F_ * F_];
  __shared__ float s_red[256];
  __shared__ float s_mu[C0_], s_rs[C0_];
  __shared__ float s_mx, s_sum;

  const int tid = threadIdx.x;
  const int bt  = blockIdx.x;                 // b*T + t
  const int b   = bt / T_;
  const int t   = bt % T_;

  if (tid < F_) s_in[tid] = in[(size_t)bt * F_ + tid];
  __syncthreads();

  // ---- conv0 (1 -> 16 ch, k=(1,3), freq pad 1) ----
  for (int e = tid; e < C0_ * F_; e += 256) {
    int c = e >> 5, f = e & 31;
    float a = (f > 0)      ? s_in[f - 1] : 0.f;
    float m = s_in[f];
    float z = (f < F_ - 1) ? s_in[f + 1] : 0.f;
    s_x0[e] = w0[c * 3 + 0] * a + w0[c * 3 + 1] * m + w0[c * 3 + 2] * z + b0[c];
  }
  __syncthreads();
  if (tid < C0_) {
    float mu = 0.f, vv = 0.f;
    for (int f = 0; f < F_; ++f) { float x = s_x0[tid * F_ + f]; mu += x; vv += x * x; }
    mu *= (1.f / F_); vv = vv * (1.f / F_) - mu * mu;
    s_mu[tid] = mu; s_rs[tid] = rsqrtf(vv + 1e-5f);
  }
  __syncthreads();
  for (int e = tid; e < C0_ * F_; e += 256) {
    int c = e >> 5, f = e & 31;
    s_x0[e] = fmaxf((s_x0[e] - s_mu[c]) * s_rs[c] * g0[f] + be0[f], 0.f);
  }
  __syncthreads();

  // ---- conv1 (16 -> 8 ch) ----
  for (int e = tid; e < C1_ * F_; e += 256) {
    int c = e >> 5, f = e & 31;
    float acc = b1[c];
    for (int ci = 0; ci < C0_; ++ci) {
      const float* wp = w1 + (c * C0_ + ci) * 3;
      float a = (f > 0)      ? s_x0[ci * F_ + f - 1] : 0.f;
      float m = s_x0[ci * F_ + f];
      float z = (f < F_ - 1) ? s_x0[ci * F_ + f + 1] : 0.f;
      acc += wp[0] * a + wp[1] * m + wp[2] * z;
    }
    s_x1[e] = acc;
  }
  __syncthreads();
  if (tid < C1_) {
    float mu = 0.f, vv = 0.f;
    for (int f = 0; f < F_; ++f) { float x = s_x1[tid * F_ + f]; mu += x; vv += x * x; }
    mu *= (1.f / F_); vv = vv * (1.f / F_) - mu * mu;
    s_mu[tid] = mu; s_rs[tid] = rsqrtf(vv + 1e-5f);
  }
  __syncthreads();
  for (int e = tid; e < C1_ * F_; e += 256) {
    int c = e >> 5, f = e & 31;
    float x = fmaxf((s_x1[e] - s_mu[c]) * s_rs[c] * g1[f] + be1[f], 0.f);
    s_x1[e] = x;
    x1out[(((size_t)b * C1_ + c) * T_ + t) * F_ + f] = x;   // x1 [B,C1,T,F]
  }
  __syncthreads();

  // ---- q/k/v convs (8 -> 64 ch) ----
  for (int e = tid; e < H_ * F_; e += 256) {
    int h = e >> 5, f = e & 31;
    float aq = qb[h], ak = kb[h], av = vb[h];
    for (int ci = 0; ci < C1_; ++ci) {
      float a = (f > 0)      ? s_x1[ci * F_ + f - 1] : 0.f;
      float m = s_x1[ci * F_ + f];
      float z = (f < F_ - 1) ? s_x1[ci * F_ + f + 1] : 0.f;
      const float* qp = qw + (h * C1_ + ci) * 3;
      const float* kp = kw + (h * C1_ + ci) * 3;
      const float* vp = vw + (h * C1_ + ci) * 3;
      aq += qp[0] * a + qp[1] * m + qp[2] * z;
      ak += kp[0] * a + kp[1] * m + kp[2] * z;
      av += vp[0] * a + vp[1] * m + vp[2] * z;
    }
    s_q[e] = aq; s_k[e] = ak; s_v[e] = av;
  }
  __syncthreads();

  // ---- w[f,g] = (q^T k)/sqrt(H) ----
  for (int e = tid; e < F_ * F_; e += 256) {
    int f = e >> 5, g = e & 31;
    float acc = 0.f;
    for (int h = 0; h < H_; ++h) acc += s_q[h * F_ + f] * s_k[h * F_ + g];
    s_w[e] = acc * 0.125f;   // 1/sqrt(64)
  }
  __syncthreads();

  // ---- joint softmax over 1024 ----
  float lm = -1e30f;
  for (int e = tid; e < F_ * F_; e += 256) lm = fmaxf(lm, s_w[e]);
  s_red[tid] = lm;
  __syncthreads();
  for (int s = 128; s > 0; s >>= 1) { if (tid < s) s_red[tid] = fmaxf(s_red[tid], s_red[tid + s]); __syncthreads(); }
  if (tid == 0) s_mx = s_red[0];
  __syncthreads();
  float ls = 0.f;
  for (int e = tid; e < F_ * F_; e += 256) { float ev = expf(s_w[e] - s_mx); s_w[e] = ev; ls += ev; }
  s_red[tid] = ls;
  __syncthreads();
  for (int s = 128; s > 0; s >>= 1) { if (tid < s) s_red[tid] += s_red[tid + s]; __syncthreads(); }
  if (tid == 0) s_sum = 1.f / s_red[0];
  __syncthreads();

  // ---- att = (w/sum) @ v^T, write gi row as f16 ----
  _Float16* girow = gi + (size_t)bt * DIN;
  const float inv = s_sum;
  for (int e = tid; e < F_ * H_; e += 256) {
    int f = e >> 6, h = e & 63;
    float acc = 0.f;
    for (int g = 0; g < F_; ++g) acc += s_w[f * F_ + g] * s_v[h * F_ + g];
    girow[f * H_ + h] = (_Float16)(acc * inv);   // gi flat index f*H + h
  }
}

// ---------------------------------------------------------------------------
// Kernel 2: f32 -> f16 convert
// ---------------------------------------------------------------------------
__global__ void cvt_f16_kernel(const float* __restrict__ s, _Float16* __restrict__ d, int n) {
  int i = blockIdx.x * 256 + threadIdx.x;
  if (i < n) d[i] = (_Float16)s[i];
}

// ---------------------------------------------------------------------------
// Kernel 3: gates = gi[32000,2048] @ wih^T[2048,768] + bih    (WMMA, f32 acc)
// One wave per 16x64 tile; K loop software-pipelined (double-buffered A + 4 B
// fragments) so v_wmma overlaps with the next K-step's global_load_b128 clause
// instead of stalling on s_wait_loadcnt 0.  grid = 3000x256.
// ---------------------------------------------------------------------------
__global__ __launch_bounds__(256)
void gates_gemm_kernel(const _Float16* __restrict__ gi, const _Float16* __restrict__ wih,
                       const float* __restrict__ bih, float* __restrict__ gates) {
  const int lane = threadIdx.x & 31;
  const int wid  = blockIdx.x * 8 + (threadIdx.x >> 5);   // global wave id
  const int mt   = wid / 12;                              // 0..1999 (M tiles of 16)
  const int ng   = wid % 12;                              // 0..11   (N groups of 64)
  const int r    = lane & 15;
  const int kh   = lane >> 4;

  const _Float16* ap  = gi  + (size_t)(mt * 16 + r) * DIN + kh * 8;
  const _Float16* bp0 = wih + (size_t)(ng * 64 +  0 + r) * DIN + kh * 8;
  const _Float16* bp1 = bp0 + (size_t)16 * DIN;
  const _Float16* bp2 = bp0 + (size_t)32 * DIN;
  const _Float16* bp3 = bp0 + (size_t)48 * DIN;

  // prologue: fragments for K-step 0
  v16h a  = ldfrag(ap);
  v16h b0 = ldfrag(bp0);
  v16h b1 = ldfrag(bp1);
  v16h b2 = ldfrag(bp2);
  v16h b3 = ldfrag(bp3);

  v8f acc0 = {}, acc1 = {}, acc2 = {}, acc3 = {};
  for (int k0 = 32; k0 < DIN; k0 += 32) {
    // issue next K-step loads first, then consume current fragments
    v16h an  = ldfrag(ap  + k0);
    v16h bn0 = ldfrag(bp0 + k0);
    v16h bn1 = ldfrag(bp1 + k0);
    v16h bn2 = ldfrag(bp2 + k0);
    v16h bn3 = ldfrag(bp3 + k0);
    acc0 = WMMA_F16(a, b0, acc0);
    acc1 = WMMA_F16(a, b1, acc1);
    acc2 = WMMA_F16(a, b2, acc2);
    acc3 = WMMA_F16(a, b3, acc3);
    a = an; b0 = bn0; b1 = bn1; b2 = bn2; b3 = bn3;
  }
  // epilogue K-step
  acc0 = WMMA_F16(a, b0, acc0);
  acc1 = WMMA_F16(a, b1, acc1);
  acc2 = WMMA_F16(a, b2, acc2);
  acc3 = WMMA_F16(a, b3, acc3);

  // C/D layout: element jj -> row mt*16 + jj + 8*kh, col ng*64 + 16*j + (lane&15)
  const int col = lane & 15;
#pragma unroll
  for (int jj = 0; jj < 8; ++jj) {
    const int m = mt * 16 + jj + kh * 8;
    float* orow = gates + (size_t)m * G3 + ng * 64 + col;
    orow[0]  = acc0[jj] + bih[ng * 64 +  0 + col];
    orow[16] = acc1[jj] + bih[ng * 64 + 16 + col];
    orow[32] = acc2[jj] + bih[ng * 64 + 32 + col];
    orow[48] = acc3[jj] + bih[ng * 64 + 48 + col];
  }
}

// ---------------------------------------------------------------------------
// Kernel 4: recurrent GRU over T=1000. 2 blocks x 16 batch rows; per step the
// 8 waves WMMA h[16,256]@whh^T[256,768] from LDS-resident f16 h (A fragments
// hoisted once per step), then f32 gate math + state update.  72 KB dyn LDS.
// ---------------------------------------------------------------------------
__global__ __launch_bounds__(256)
void gru_main_kernel(const float* __restrict__ gates, const _Float16* __restrict__ whh,
                     const float* __restrict__ bhh, float* __restrict__ o,
                     float* __restrict__ h_att_out) {
  extern __shared__ char smem[];
  _Float16* h16 = (_Float16*)smem;                              // [16][256] f16
  float*    hf  = (float*)(smem + 16 * DH * 2);                 // [16][256] f32
  float*    gh  = (float*)(smem + 16 * DH * 2 + 16 * DH * 4);   // [16][768] f32

  const int tid  = threadIdx.x;
  const int lane = tid & 31;
  const int wave = tid >> 5;
  const int b0   = blockIdx.x * 16;
  const int r    = lane & 15;
  const int kh   = lane >> 4;
  const int col  = lane & 15;

  for (int e = tid; e < 16 * DH; e += 256) { hf[e] = 0.f; h16[e] = (_Float16)0.f; }
  __syncthreads();

  const _Float16* arow = h16 + r * DH + kh * 8;

  for (int t = 0; t < T_; ++t) {
    // warm L2/L1 for next step's input-gate rows while we do this step's WMMAs
    if (t + 1 < T_)
      __builtin_prefetch(gates + ((size_t)(b0 + (tid & 15)) * T_ + (t + 1)) * G3 + (tid >> 4) * 48, 0, 1);

    // hoist the 8 A fragments (h state) once per step
    v16h afr[8];
#pragma unroll
    for (int kt = 0; kt < 8; ++kt) afr[kt] = ldfrag(arow + kt * 32);

#pragma unroll
    for (int i = 0; i < 6; ++i) {
      const int nt = wave * 6 + i;                              // 48 N-tiles total
      const _Float16* brow = whh + (size_t)(nt * 16 + r) * DH + kh * 8;
      // clause all 8 B fragments, then 8 back-to-back WMMAs
      v16h bfr[8];
#pragma unroll
      for (int kt = 0; kt < 8; ++kt) bfr[kt] = ldfrag(brow + kt * 32);
      v8f acc = {};
#pragma unroll
      for (int kt = 0; kt < 8; ++kt) acc = WMMA_F16(afr[kt], bfr[kt], acc);
#pragma unroll
      for (int jj = 0; jj < 8; ++jj)
        gh[(jj + kh * 8) * G3 + nt * 16 + col] = acc[jj];
    }
    __syncthreads();

    // gate nonlinearities + state update (f32)
    for (int e = tid; e < 16 * DH; e += 256) {
      const int m = e >> 8, c = e & 255;
      const float* grow = gates + ((size_t)(b0 + m) * T_ + t) * G3;
      float ir  = grow[c];
      float iz  = grow[DH + c];
      float in_ = grow[2 * DH + c];
      float hr = gh[m * G3 + c]          + bhh[c];
      float hz = gh[m * G3 + DH + c]     + bhh[DH + c];
      float hn = gh[m * G3 + 2 * DH + c] + bhh[2 * DH + c];
      float rr = sigmoidf_(ir + hr);
      float zz = sigmoidf_(iz + hz);
      float nn = tanhf(in_ + rr * hn);
      float hv = (1.f - zz) * nn + zz * hf[e];
      hf[e]  = hv;
      h16[e] = (_Float16)hv;
      o[((size_t)(b0 + m) * T_ + t) * DH + c] = hv;
    }
    __syncthreads();
  }

  for (int e = tid; e < 16 * DH; e += 256) {
    const int m = e >> 8, c = e & 255;
    h_att_out[(b0 + m) * DH + c] = hf[e];    // h_atten [1,B,256]
  }
}

// ---------------------------------------------------------------------------
// Kernel 5: gru_snr scans over dim0 == B (32 steps), batch == T (1000 lanes).
// One 64-thread block per t.  Also computes lsnr.
// ---------------------------------------------------------------------------
__global__ __launch_bounds__(64)
void gru_snr_kernel(const float* __restrict__ o,
                    const float* __restrict__ wih, const float* __restrict__ whh,
                    const float* __restrict__ bih, const float* __restrict__ bhh,
                    const float* __restrict__ fcw, const float* __restrict__ fcb,
                    float* __restrict__ lsnr, float* __restrict__ h_snr_out) {
  __shared__ float xr[DH];
  __shared__ float hh[16];
  __shared__ float g_i[48], g_h[48];
  const int tid = threadIdx.x;
  const int t   = blockIdx.x;

  if (tid < 16) hh[tid] = 0.f;
  __syncthreads();

  for (int s = 0; s < B_; ++s) {
    const float* orow = o + ((size_t)s * T_ + t) * DH;
    for (int e = tid; e < DH; e += 64) xr[e] = orow[e];
    __syncthreads();
    if (tid < 48) {
      float a = bih[tid];
      const float* wp = wih + tid * DH;
      for (int k = 0; k < DH; ++k) a += wp[k] * xr[k];
      g_i[tid] = a;
      float b = bhh[tid];
      const float* vp = whh + tid * 16;
      for (int k = 0; k < 16; ++k) b += vp[k] * hh[k];
      g_h[tid] = b;
    }
    __syncthreads();
    if (tid < 16) {
      float rr = sigmoidf_(g_i[tid]      + g_h[tid]);
      float zz = sigmoidf_(g_i[16 + tid] + g_h[16 + tid]);
      float nn = tanhf(g_i[32 + tid] + rr * g_h[32 + tid]);
      hh[tid] = (1.f - zz) * nn + zz * hh[tid];
    }
    __syncthreads();
    if (tid == 0) {
      float a = fcb[0];
      for (int k = 0; k < 16; ++k) a += fcw[k] * hh[k];
      lsnr[s * T_ + t] = sigmoidf_(a) * 50.f - 15.f;   // lsnr [B,T,1]
    }
    __syncthreads();
  }
  if (tid < 16) h_snr_out[t * 16 + tid] = hh[tid];     // h_snr [1,T,16]
}

// ---------------------------------------------------------------------------
// Kernel 6: xe = LN(x1 + e); m = sigmoid(LN(conv2(xe)))
// ---------------------------------------------------------------------------
__global__ __launch_bounds__(64)
void epilogue_kernel(const float* __restrict__ x1, const float* __restrict__ o,
                     const float* __restrict__ lng, const float* __restrict__ lnb,
                     const float* __restrict__ c2w, const float* __restrict__ c2b,
                     const float* __restrict__ g2, const float* __restrict__ b2,
                     float* __restrict__ m_out) {
  __shared__ float xe[C1_ * F_];
  __shared__ float mrow[F_];
  __shared__ float smu, srs;
  const int tid = threadIdx.x;
  const int bt  = blockIdx.x;
  const int b   = bt / T_, t = bt % T_;

  for (int e = tid; e < C1_ * F_; e += 64) {
    int c = e >> 5, f = e & 31;
    float xv = x1[(((size_t)b * C1_ + c) * T_ + t) * F_ + f];
    float ev = o[((size_t)b * T_ + t) * DH + c * F_ + f];   // e[b,c,t,f] = o[b,t,c*F+f]
    xe[e] = xv + ev;
  }
  __syncthreads();
  if (tid < C1_) {
    float mu = 0.f, vv = 0.f;
    for (int f = 0; f < F_; ++f) { float x = xe[tid * F_ + f]; mu += x; vv += x * x; }
    mu *= (1.f / F_); vv = vv * (1.f / F_) - mu * mu;
    float rs = rsqrtf(vv + 1e-5f);
    for (int f = 0; f < F_; ++f) xe[tid * F_ + f] = (xe[tid * F_ + f] - mu) * rs * lng[f] + lnb[f];
  }
  __syncthreads();
  if (tid < F_) {
    const int f = tid;
    float acc = c2b[0];
    for (int c = 0; c < C1_; ++c) {
      const float* wp = c2w + c * 3;
      float a = (f > 0)      ? xe[c * F_ + f - 1] : 0.f;
      float m = xe[c * F_ + f];
      float z = (f < F_ - 1) ? xe[c * F_ + f + 1] : 0.f;
      acc += wp[0] * a + wp[1] * m + wp[2] * z;
    }
    mrow[f] = acc;
  }
  __syncthreads();
  if (tid == 0) {
    float mu = 0.f, vv = 0.f;
    for (int f = 0; f < F_; ++f) { mu += mrow[f]; vv += mrow[f] * mrow[f]; }
    mu *= (1.f / F_); vv = vv * (1.f / F_) - mu * mu;
    smu = mu; srs = rsqrtf(vv + 1e-5f);
  }
  __syncthreads();
  if (tid < F_) {
    float x = (mrow[tid] - smu) * srs * g2[tid] + b2[tid];
    m_out[(size_t)bt * F_ + tid] = sigmoidf_(x);           // m [B,1,T,F]
  }
}

// ---------------------------------------------------------------------------
extern "C" void kernel_launch(void* const* d_in, const int* in_sizes, int n_in,
                              void* d_out, int out_size, void* d_ws, size_t ws_size,
                              hipStream_t stream) {
  const float* input    = (const float*)d_in[0];
  const float* conv0_w  = (const float*)d_in[1];
  const float* conv0_b  = (const float*)d_in[2];
  const float* ln0_g    = (const float*)d_in[3];
  const float* ln0_b    = (const float*)d_in[4];
  const float* conv1_w  = (const float*)d_in[5];
  const float* conv1_b  = (const float*)d_in[6];
  const float* ln1_g    = (const float*)d_in[7];
  const float* ln1_b    = (const float*)d_in[8];
  const float* q_w      = (const float*)d_in[9];
  const float* q_b      = (const float*)d_in[10];
  const float* k_w      = (const float*)d_in[11];
  const float* k_b      = (const float*)d_in[12];
  const float* v_w      = (const float*)d_in[13];
  const float* v_b      = (const float*)d_in[14];
  const float* gruo_wih = (const float*)d_in[15];
  const float* gruo_whh = (const float*)d_in[16];
  const float* gruo_bih = (const float*)d_in[17];
  const float* gruo_bhh = (const float*)d_in[18];
  const float* ln_g     = (const float*)d_in[19];
  const float* ln_b     = (const float*)d_in[20];
  const float* grus_wih = (const float*)d_in[21];
  const float* grus_whh = (const float*)d_in[22];
  const float* grus_bih = (const float*)d_in[23];
  const float* grus_bhh = (const float*)d_in[24];
  const float* fc_w     = (const float*)d_in[25];
  const float* fc_b     = (const float*)d_in[26];
  const float* conv2_w  = (const float*)d_in[27];
  const float* conv2_b  = (const float*)d_in[28];
  const float* ln2_g    = (const float*)d_in[29];
  const float* ln2_b    = (const float*)d_in[30];

  // ---- workspace carve-out (~300 MB total) ----
  char* ws = (char*)d_ws;
  size_t off = 0;
  auto carve = [&](size_t bytes) -> void* {
    void* p = ws + off;
    off += (bytes + 255) & ~(size_t)255;
    return p;
  };
  float*    x1buf = (float*)   carve((size_t)B_ * C1_ * T_ * F_ * sizeof(float));  // 32.8 MB
  _Float16* gi_h  = (_Float16*)carve((size_t)M_ * DIN * sizeof(_Float16));         // 131 MB
  _Float16* wih_h = (_Float16*)carve((size_t)G3 * DIN * sizeof(_Float16));         // 3.1 MB
  _Float16* whh_h = (_Float16*)carve((size_t)G3 * DH * sizeof(_Float16));          // 0.4 MB
  float*    gates = (float*)   carve((size_t)M_ * G3 * sizeof(float));             // 98.3 MB
  float*    obuf  = (float*)   carve((size_t)M_ * DH * sizeof(float));             // 32.8 MB
  (void)ws_size; (void)in_sizes; (void)n_in; (void)out_size;

  // ---- output packing (f32, concatenated in return order) ----
  float* out   = (float*)d_out;
  float* m_out = out;                                 // [B,1,T,F]  = 1,024,000
  float* lsnr  = out + (size_t)B_ * T_ * F_;          // [B,T,1]    =    32,000
  float* h_att = lsnr + (size_t)B_ * T_;              // [1,B,256]  =     8,192
  float* h_snr = h_att + (size_t)B_ * DH;             // [1,T,16]   =    16,000

  // 1) weight converts
  cvt_f16_kernel<<<(G3 * DIN + 255) / 256, 256, 0, stream>>>(gruo_wih, wih_h, G3 * DIN);
  cvt_f16_kernel<<<(G3 * DH + 255) / 256, 256, 0, stream>>>(gruo_whh, whh_h, G3 * DH);

  // 2) fused conv/LN/attention frontend -> x1, gi(f16)
  frontend_kernel<<<M_, 256, 0, stream>>>(input, conv0_w, conv0_b, ln0_g, ln0_b,
                                          conv1_w, conv1_b, ln1_g, ln1_b,
                                          q_w, q_b, k_w, k_b, v_w, v_b,
                                          x1buf, gi_h);

  // 3) big WMMA GEMM: gates = gi @ wih^T + bih  (M=32000,N=768,K=2048)
  gates_gemm_kernel<<<(M_ / 16) * (G3 / 64) / 8, 256, 0, stream>>>(gi_h, wih_h, gruo_bih, gates);

  // 4) recurrent GRU (WMMA per step), writes o and h_atten
  gru_main_kernel<<<2, 256, 16 * DH * 2 + 16 * DH * 4 + 16 * G3 * 4, stream>>>(
      gates, whh_h, gruo_bhh, obuf, h_att);

  // 5) gru_snr + lsnr
  gru_snr_kernel<<<T_, 64, 0, stream>>>(obuf, grus_wih, grus_whh, grus_bih, grus_bhh,
                                        fc_w, fc_b, lsnr, h_snr);

  // 6) epilogue -> m
  epilogue_kernel<<<M_, 64, 0, stream>>>(x1buf, obuf, ln_g, ln_b, conv2_w, conv2_b,
                                         ln2_g, ln2_b, m_out);
}
